// WindowAttention_82265803587927
// MI455X (gfx1250) — matrix operand order repaired and checked by reading, not compiled, not run on hardware
//
#include <hip/hip_runtime.h>
#include <hip/hip_bf16.h>
#include <math.h>

typedef _Float16 half_t;
typedef __attribute__((ext_vector_type(16))) _Float16 v16h;
typedef __attribute__((ext_vector_type(8)))  _Float16 v8h;
typedef __attribute__((ext_vector_type(8)))  float    v8f;
typedef __attribute__((ext_vector_type(4))) unsigned int u32x4;
typedef __attribute__((ext_vector_type(8))) int          i32x8;
typedef __attribute__((ext_vector_type(4))) int          i32x4;

// ---------------------------------------------------------------------------
// WMMA helpers (gfx1250, wave32). D = A(16x32 f16) * B(32x16 f16) + C(16x16 f32)
// ---------------------------------------------------------------------------
__device__ __forceinline__ v8f wmma_f16(v16h a, v16h b, v8f c) {
  return __builtin_amdgcn_wmma_f32_16x16x32_f16(
      /*neg_a=*/false, a, /*neg_b=*/false, b,
      /*c_mod=*/(short)0, c, /*reuse_a=*/false, /*reuse_b=*/false);
}

// A fragment from row-major [16][ld] tile (ld in halves).
// lane m=lane&15; kb = lane<16 ? 0 : 8; halves 0..7 = K kb..kb+7,
// halves 8..15 = K kb+16..kb+23.
__device__ __forceinline__ v16h load_a_frag(const half_t* base, int ld) {
  const int lane = threadIdx.x & 31;
  const int m  = lane & 15;
  const int kb = (lane < 16) ? 0 : 8;
  const half_t* p = base + m * ld + kb;
  v8h lo = *(const v8h*)p;
  v8h hi = *(const v8h*)(p + 16);
  return __builtin_shufflevector(lo, hi, 0,1,2,3,4,5,6,7,8,9,10,11,12,13,14,15);
}

// B fragment from B^T stored row-major [16][ld] (row n = output col, cols = K).
// lane n=lane&15; kb = lane<16 ? 0 : 8; halves 0..7 = K kb..kb+7,
// halves 8..15 = K kb+16..kb+23.
__device__ __forceinline__ v16h load_b_frag(const half_t* base, int ld) {
  const int lane = threadIdx.x & 31;
  const int n  = lane & 15;
  const int kb = (lane < 16) ? 0 : 8;
  const half_t* p = base + n * ld + kb;
  v8h lo = *(const v8h*)p;
  v8h hi = *(const v8h*)(p + 16);
  return __builtin_shufflevector(lo, hi, 0,1,2,3,4,5,6,7,8,9,10,11,12,13,14,15);
}

// ---------------------------------------------------------------------------
// Tensor Data Mover: async 2D f16 tile load Global -> LDS.
// D# layout per CDNA5 ISA (groups 0/1; 2D so groups 2/3
// are zero, tile_dim2=0 = unused). Reads past tensor_dim return ZERO, which we
// exploit for 49->64 row padding. Tracked by TENSORcnt.
// This toolchain exposes the 6-arg builtin (extra zero-filled int32x8 group).
// ---------------------------------------------------------------------------
__device__ __forceinline__ void tdm_load_2d(unsigned lds_addr, const void* gptr,
                                            unsigned tile_cols, unsigned tile_rows,
                                            unsigned tensor_rows, unsigned row_stride) {
  unsigned long long ga = (unsigned long long)gptr;
  u32x4 g0;
  g0[0] = 1u;                                               // count=1 (valid), user mode
  g0[1] = lds_addr;                                         // LDS byte address
  g0[2] = (unsigned)ga;                                     // global addr [31:0]
  g0[3] = (unsigned)((ga >> 32) & 0x01FFFFFFu) | (2u << 30);// addr[56:32] | type=2
  i32x8 g1;
  g1[0] = 0x00010000;                                       // data_size=1 (2 bytes)
  g1[1] = (int)((tile_cols & 0xFFFFu) << 16);               // tensor_dim0 lo16 (==tile_cols)
  g1[2] = (int)((tensor_rows & 0xFFFFu) << 16);             // tensor_dim0 hi=0 | tensor_dim1 lo16
  g1[3] = (int)((tile_cols & 0xFFFFu) << 16);               // tensor_dim1 hi=0 | tile_dim0
  g1[4] = (int)(tile_rows & 0xFFFFu);                       // tile_dim1 | tile_dim2=0
  g1[5] = (int)row_stride;                                  // tensor_dim0_stride lo32
  g1[6] = 0;                                                // stride hi | dim1_stride lo
  g1[7] = 0;
  i32x4 gz4 = {0, 0, 0, 0};
  i32x8 gz8 = {0, 0, 0, 0, 0, 0, 0, 0};
  __builtin_amdgcn_tensor_load_to_lds(g0, g1, gz4, gz4, gz8, 0);
}

// ---------------------------------------------------------------------------
// Prep kernels
// ---------------------------------------------------------------------------
__global__ void transpose_to_half(const float* __restrict__ W,
                                  half_t* __restrict__ Wt, int K, int N) {
  int idx = blockIdx.x * 256 + threadIdx.x;   // idx = n*K + k
  if (idx < K * N) {
    int n = idx / K, k = idx - n * K;
    Wt[idx] = (half_t)W[(size_t)k * N + n];
  }
}

__global__ void prep_scales(const float* __restrict__ ls, float* __restrict__ scales) {
  int i = threadIdx.x;
  if (i < 8) scales[i] = __expf(fminf(fmaxf(ls[i], -100.0f), 4.6051702f)); // log(100)
}

__device__ __forceinline__ float cpb_coord(int d) {
  float v = (float)d * (8.0f / 6.0f);
  float s = (v > 0.f) ? 1.f : ((v < 0.f) ? -1.f : 0.f);
  return s * logf(fabsf(v) + 1.0f) * 0.48089835f;   // 1/log(8)
}

__global__ void prep_cpb(const float* __restrict__ w1, const float* __restrict__ b1,
                         const float* __restrict__ w2, float* __restrict__ bias_table) {
  int i = blockIdx.x * blockDim.x + threadIdx.x;    // 0..168 = th*13+tw
  if (i >= 169) return;
  float t0 = cpb_coord(i / 13 - 6);
  float t1 = cpb_coord(i % 13 - 6);
  float acc[8] = {0.f,0.f,0.f,0.f,0.f,0.f,0.f,0.f};
  for (int j = 0; j < 512; ++j) {
    float pre = t0 * w1[j] + t1 * w1[512 + j] + b1[j];
    float g = 0.5f * pre * (1.0f + erff(pre * 0.70710678f));  // exact gelu
#pragma unroll
    for (int h = 0; h < 8; ++h) acc[h] += g * w2[j * 8 + h];
  }
#pragma unroll
  for (int h = 0; h < 8; ++h) bias_table[i * 8 + h] = acc[h];
}

__global__ void prep_bias(const float* __restrict__ bias_table, float* __restrict__ bias) {
  int idx = blockIdx.x * 256 + threadIdx.x;         // h*2401 + i*49 + j
  if (idx >= 8 * 49 * 49) return;
  int h = idx / 2401, rem = idx - h * 2401;
  int i = rem / 49, j = rem - i * 49;
  int dh = i / 7 - j / 7 + 6;
  int dw = i % 7 - j % 7 + 6;
  float t = bias_table[(dh * 13 + dw) * 8 + h];
  bias[idx] = 16.0f / (1.0f + __expf(-t));
}

// ---------------------------------------------------------------------------
// Tiled WMMA GEMM, K=256 fixed:  C[M][Nout] = A[M][256] @ Bt^T + bias
// Bt pre-transposed f16 [Nout][256]. Block = 256 threads, tile 64x128.
// B tile (and A tile when already f16) arrives via TDM async DMA; f32 A tile
// is converted on the VALU concurrently with the DMA. 8 waves, each computes a
// 16x64 strip: 4 accumulators, 32 WMMAs, A-fragment reused 4x.
// ---------------------------------------------------------------------------
template <typename AT, typename OutT>
__global__ __launch_bounds__(256) void gemm_wmma_k256(
    const AT* __restrict__ A, const half_t* __restrict__ Bt,
    const float* __restrict__ bias, OutT* __restrict__ C, int M, int Nout) {
  __shared__ alignas(32) half_t As [64][256];    // 32 KB
  __shared__ alignas(32) half_t Bts[128][256];   // 64 KB
  const int m0 = blockIdx.x * 64;
  const int n0 = blockIdx.y * 128;
  const int tid  = threadIdx.x;
  const int wave = tid >> 5;
  const int lane = tid & 31;

  // Async DMA of the B tile (128x256 halves, contiguous rows).
  if (wave == 0) {
    tdm_load_2d((unsigned)(size_t)&Bts[0][0], Bt + (size_t)n0 * 256,
                /*cols*/256, /*rows*/128, /*tensor_rows*/128, /*stride*/256);
  }

  if constexpr (sizeof(AT) == 2) {
    // A already f16: DMA it too.
    if (wave == 0) {
      tdm_load_2d((unsigned)(size_t)&As[0][0], A + (size_t)m0 * 256,
                  256, 64, 64, 256);
    }
  } else {
    // f32 -> f16 conversion on the VALU, overlapped with the TDM transfer.
    for (int t = tid; t < 64 * 32; t += 256) {
      int r = t >> 5, c = (t & 31) << 3;
      const AT* src = A + (size_t)(m0 + r) * 256 + c;
      v8h hv;
#pragma unroll
      for (int e = 0; e < 8; ++e) hv[e] = (half_t)(float)src[e];
      *(v8h*)&As[r][c] = hv;
    }
  }
  __builtin_amdgcn_s_wait_tensorcnt(0);
  __syncthreads();

  const int mt = wave & 3;            // 16-row strip
  const int ng = (wave >> 2) * 4;     // first of four 16-col tiles
  v8f acc[4] = {};
#pragma unroll
  for (int ks = 0; ks < 8; ++ks) {    // K = 256 = 8 * 32
    v16h a = load_a_frag(&As[mt * 16][0] + ks * 32, 256);
#pragma unroll
    for (int j = 0; j < 4; ++j) {
      v16h b = load_b_frag(&Bts[(ng + j) * 16][0] + ks * 32, 256);
      acc[j] = wmma_f16(a, b, acc[j]);
    }
  }

  const int n  = lane & 15;
  const int mb = (lane < 16) ? 0 : 8;
#pragma unroll
  for (int j = 0; j < 4; ++j) {
    int gn = n0 + (ng + j) * 16 + n;
    float bv = bias[gn];
#pragma unroll
    for (int r = 0; r < 8; ++r) {
      int gm = m0 + mt * 16 + mb + r;
      if (gm < M) C[(size_t)gm * Nout + gn] = (OutT)(acc[j][r] + bv);
    }
  }
}

// ---------------------------------------------------------------------------
// Attention: one wave per (window b, head h). 49 tokens padded to 64.
// Q/K tiles arrive via TDM (tensor_dim1=49 -> rows 49..63 auto-zeroed);
// V is loaded coalesced and transposed into LDS while the DMA runs.
// ---------------------------------------------------------------------------
__global__ __launch_bounds__(32) void attn_kernel(
    const half_t* __restrict__ qkv, const float* __restrict__ scales,
    const float* __restrict__ bias, half_t* __restrict__ out) {
  const int b = blockIdx.x >> 3;
  const int h = blockIdx.x & 7;
  const int lane = threadIdx.x;

  __shared__ alignas(32) half_t Qs[64][32];   // 4 KB
  __shared__ alignas(32) half_t Ks[64][32];   // 4 KB
  __shared__ alignas(32) half_t Vt[32][64];   // 4 KB (V transposed: [d][m])
  __shared__ alignas(32) float  Sm[64][64];   // 16 KB
  __shared__ alignas(32) half_t Ps[64][64];   // 8 KB

  const size_t rowbase = (size_t)b * 49;
  const half_t* qbase = qkv + rowbase * 768 + h * 32;

  // Async Q/K tiles: 32 cols x 64 rows, only 49 rows valid, row stride 768.
  tdm_load_2d((unsigned)(size_t)&Qs[0][0], qbase,       32, 64, 49, 768);
  tdm_load_2d((unsigned)(size_t)&Ks[0][0], qbase + 256, 32, 64, 49, 768);

  // V: coalesced global reads (lane = head-dim), transposed store.
  for (int m = 0; m < 49; ++m) Vt[lane][m] = qbase[(size_t)m * 768 + 512 + lane];
  for (int m = 49; m < 64; ++m) Vt[lane][m] = (half_t)0.f;

  __builtin_amdgcn_s_wait_tensorcnt(0);
  __syncthreads();

  // Cosine-attn: L2-normalize q (logit scale folded in) and k rows.
  const float scale = scales[h];
  for (int rr = 0; rr < 2; ++rr) {
    int r = lane + rr * 32;
    if (r < 49) {
      float sq = 0.f, sk = 0.f;
      for (int d = 0; d < 32; ++d) {
        float q = (float)Qs[r][d], k = (float)Ks[r][d];
        sq += q * q; sk += k * k;
      }
      float iq = scale * rsqrtf(sq + 1e-12f);
      float ik = rsqrtf(sk + 1e-12f);
      for (int d = 0; d < 32; ++d) {
        Qs[r][d] = (half_t)((float)Qs[r][d] * iq);
        Ks[r][d] = (half_t)((float)Ks[r][d] * ik);
      }
    }
  }
  __syncthreads();

  const int n  = lane & 15;
  const int mb = (lane < 16) ? 0 : 8;

  // S = Qn @ Kn^T  (Bt for this product is Ks itself).
#pragma unroll
  for (int mt = 0; mt < 4; ++mt) {
    v16h a = load_a_frag(&Qs[mt * 16][0], 32);
#pragma unroll
    for (int nt = 0; nt < 4; ++nt) {
      v16h bf = load_b_frag(&Ks[nt * 16][0], 32);
      v8f c = {};
      c = wmma_f16(a, bf, c);
#pragma unroll
      for (int r = 0; r < 8; ++r) Sm[mt * 16 + mb + r][nt * 16 + n] = c[r];
    }
  }
  __syncthreads();

  // Softmax with CPB bias over cols 0..48; build P (f16) with zero padding.
  const float* bh = bias + (size_t)h * 49 * 49;
  for (int rr = 0; rr < 2; ++rr) {
    int r = lane + rr * 32;
    if (r < 49) {
      float mx = -1e30f;
      for (int j = 0; j < 49; ++j) mx = fmaxf(mx, Sm[r][j] + bh[r * 49 + j]);
      float s = 0.f;
      for (int j = 0; j < 49; ++j) {
        float e = __expf(Sm[r][j] + bh[r * 49 + j] - mx);
        Sm[r][j] = e; s += e;
      }
      float inv = 1.f / s;
      for (int j = 0; j < 49; ++j) Ps[r][j] = (half_t)(Sm[r][j] * inv);
      for (int j = 49; j < 64; ++j) Ps[r][j] = (half_t)0.f;
    } else {
      for (int j = 0; j < 64; ++j) Ps[r][j] = (half_t)0.f;
    }
  }
  __syncthreads();

  // O = P @ V  (Bt for this product is Vt: Bt[n=d][k=m] = V[m][d]).
#pragma unroll
  for (int mt = 0; mt < 4; ++mt) {
#pragma unroll
    for (int nt = 0; nt < 2; ++nt) {
      v8f c = {};
#pragma unroll
      for (int ks = 0; ks < 2; ++ks) {
        v16h a  = load_a_frag(&Ps[mt * 16][0] + ks * 32, 64);
        v16h bf = load_b_frag(&Vt[nt * 16][0] + ks * 32, 64);
        c = wmma_f16(a, bf, c);
      }
#pragma unroll
      for (int r = 0; r < 8; ++r) {
        int m = mt * 16 + mb + r;
        if (m < 49)
          out[(rowbase + m) * 256 + h * 32 + nt * 16 + n] = (half_t)c[r];
      }
    }
  }
}

// ---------------------------------------------------------------------------
// Host launcher
// ---------------------------------------------------------------------------
extern "C" void kernel_launch(void* const* d_in, const int* in_sizes, int n_in,
                              void* d_out, int out_size, void* d_ws, size_t ws_size,
                              hipStream_t stream) {
  const float* x           = (const float*)d_in[0];
  const float* qkv_w       = (const float*)d_in[1];
  const float* qkv_b       = (const float*)d_in[2];
  const float* proj_w      = (const float*)d_in[3];
  const float* proj_b      = (const float*)d_in[4];
  const float* logit_scale = (const float*)d_in[5];
  const float* cpb_w1      = (const float*)d_in[6];
  const float* cpb_b1      = (const float*)d_in[7];
  const float* cpb_w2      = (const float*)d_in[8];
  float* out = (float*)d_out;

  const int M    = in_sizes[0] / 256;   // B * 49 = 100352
  const int Bwin = M / 49;              // 2048

  size_t off = 0;
  auto alloc = [&](size_t bytes) {
    size_t p = off;
    off += (bytes + 255) & ~(size_t)255;
    return p;
  };
  char* wsb = (char*)d_ws;
  float*  scales   = (float*) (wsb + alloc(8 * sizeof(float)));
  float*  bias_tab = (float*) (wsb + alloc(169 * 8 * sizeof(float)));
  float*  biasf    = (float*) (wsb + alloc(8 * 49 * 49 * sizeof(float)));
  half_t* Wqkv     = (half_t*)(wsb + alloc((size_t)768 * 256 * sizeof(half_t)));
  half_t* Wproj    = (half_t*)(wsb + alloc((size_t)256 * 256 * sizeof(half_t)));
  half_t* qkvh     = (half_t*)(wsb + alloc((size_t)M * 768 * sizeof(half_t)));
  half_t* attnh    = (half_t*)(wsb + alloc((size_t)M * 256 * sizeof(half_t)));

  transpose_to_half<<<(768 * 256 + 255) / 256, 256, 0, stream>>>(qkv_w, Wqkv, 256, 768);
  transpose_to_half<<<(256 * 256 + 255) / 256, 256, 0, stream>>>(proj_w, Wproj, 256, 256);
  prep_scales<<<1, 32, 0, stream>>>(logit_scale, scales);
  prep_cpb<<<1, 256, 0, stream>>>(cpb_w1, cpb_b1, cpb_w2, bias_tab);
  prep_bias<<<(8 * 49 * 49 + 255) / 256, 256, 0, stream>>>(bias_tab, biasf);

  // QKV GEMM: [M,256] x [256,768] -> f16 [M,768]
  gemm_wmma_k256<float, half_t>
      <<<dim3((M + 63) / 64, 768 / 128), 256, 0, stream>>>(x, Wqkv, qkv_b, qkvh, M, 768);

  // Per-(window, head) attention.
  attn_kernel<<<Bwin * 8, 32, 0, stream>>>(qkvh, scales, biasf, attnh);

  // Output projection: [M,256] x [256,256] -> f32 d_out
  gemm_wmma_k256<half_t, float>
      <<<dim3((M + 63) / 64, 256 / 128), 256, 0, stream>>>(attnh, Wproj, proj_b, out, M, 256);
}
